// LSTMAutoencoder_72258529788617
// MI455X (gfx1250) — matrix operand order repaired and checked by reading, not compiled, not run on hardware
//
#include <hip/hip_runtime.h>
#include <hip/hip_bf16.h>

// ---------------------------------------------------------------------------
// LSTM autoencoder on gfx1250 (MI455X): bf16 WMMA recurrence.
// One fused kernel per timestep runs both LSTM layers (+ output projection in
// the decoder). Each workgroup = 32 batch rows x all 1024 gate columns
// (2 row-blocks of 16 -> each loaded B fragment feeds two v_wmma, halving L2
// weight traffic vs a 16-row tile). 8 waves; each wave owns a 32-col H-slice
// across all four gates so sigmoid/tanh runs entirely in accumulator regs.
// ---------------------------------------------------------------------------

typedef __attribute__((ext_vector_type(16))) __bf16 bf16x16;
typedef __attribute__((ext_vector_type(8)))  float  f32x8;

constexpr int Bsz = 2048;
constexpr int Ssz = 128;
constexpr int Dsz = 128;
constexpr int Hsz = 256;
constexpr int Gsz = 1024;   // 4*H
constexpr int BT  = 32;     // batch tile rows per workgroup

union FragBF {
    bf16x16        v;
    uint4          q[2];
    unsigned short h[16];
};

__device__ __forceinline__ unsigned short bf16bits(float f) {
    unsigned int u = __float_as_uint(f);
    u += 0x7FFFu + ((u >> 16) & 1u);   // round-to-nearest-even
    return (unsigned short)(u >> 16);
}

__device__ __forceinline__ float sigmoidf_(float x) {
    return 1.0f / (1.0f + __expf(-x));
}
__device__ __forceinline__ float tanhf_(float x) {
    return __builtin_fmaf(2.0f, sigmoidf_(2.0f * x), -1.0f);
}

// A-fragment (16x32 bf16, M x K), ISA 7.12.2 layout:
// lanes 0-15: K = [0..7] & [16..23]; lanes 16-31: K = [8..15] & [24..31]
__device__ __forceinline__ FragBF load_a_bf16(const unsigned short* base, int ld,
                                              int row, int kc, int lane) {
    FragBF f;
    const unsigned short* p = base + (size_t)row * ld + kc + ((lane >> 4) * 8);
    f.q[0] = *(const uint4*)(p);
    f.q[1] = *(const uint4*)(p + 16);
    return f;
}

// A-fragment from fp32 source with on-the-fly RNE bf16 conversion.
__device__ __forceinline__ FragBF load_a_f32cvt(const float* row_ptr, int kc, int lane) {
    FragBF a;
    const int off = kc + ((lane >> 4) * 8);
    const float4 u0 = *(const float4*)(row_ptr + off);
    const float4 u1 = *(const float4*)(row_ptr + off + 4);
    const float4 u2 = *(const float4*)(row_ptr + off + 16);
    const float4 u3 = *(const float4*)(row_ptr + off + 20);
    a.h[0]  = bf16bits(u0.x); a.h[1]  = bf16bits(u0.y);
    a.h[2]  = bf16bits(u0.z); a.h[3]  = bf16bits(u0.w);
    a.h[4]  = bf16bits(u1.x); a.h[5]  = bf16bits(u1.y);
    a.h[6]  = bf16bits(u1.z); a.h[7]  = bf16bits(u1.w);
    a.h[8]  = bf16bits(u2.x); a.h[9]  = bf16bits(u2.y);
    a.h[10] = bf16bits(u2.z); a.h[11] = bf16bits(u2.w);
    a.h[12] = bf16bits(u3.x); a.h[13] = bf16bits(u3.y);
    a.h[14] = bf16bits(u3.z); a.h[15] = bf16bits(u3.w);
    return a;
}

// One K-chunk (32) of the gate GEMM for all 8 owned N-tiles and both
// 16-row blocks. W is row-major [N, K]; each lane reads K = [0..7]&[16..23] (lanes 16-31: +8)
// of its column n (W^T needs no transpose).
__device__ __forceinline__ void mm_step2(f32x8 acc[2][8], const FragBF& a0,
                                         const FragBF& a1,
                                         const unsigned short* __restrict__ W,
                                         int ldk, int kc, int w, int lane) {
    const int arow = lane & 15;
    const int koff = kc + ((lane >> 4) * 8);
#pragma unroll
    for (int j = 0; j < 8; ++j) {
        const int ncol = (j >> 1) * 256 + w * 32 + (j & 1) * 16 + arow;
        FragBF b;
        const unsigned short* p = W + (size_t)ncol * ldk + koff;
        b.q[0] = *(const uint4*)(p);
        b.q[1] = *(const uint4*)(p + 16);
        acc[0][j] = __builtin_amdgcn_wmma_f32_16x16x32_bf16(
            false, a0.v, false, b.v, (short)0, acc[0][j], false, false);
        acc[1][j] = __builtin_amdgcn_wmma_f32_16x16x32_bf16(
            false, a1.v, false, b.v, (short)0, acc[1][j], false, false);
    }
}

// In-register LSTM gate nonlinearity for this wave's 32x32 slice.
// acc[rb][j]: rb = 16-row block, gate = j>>1 (i,f,g,o), sub-tile p = j&1.
__device__ __forceinline__ void cell_elemwise(f32x8 acc[2][8],
                                              const float* __restrict__ bias,
                                              float* cbuf, unsigned short* hbuf_g,
                                              unsigned short* hlds,
                                              int rowbase, int w, int lane) {
    const int arow  = lane & 15;
    const int rhalf = (lane >> 4) * 8;
#pragma unroll
    for (int pp = 0; pp < 2; ++pp) {
        const int hcol = w * 32 + pp * 16 + arow;
        const float bi = bias[hcol];
        const float bf = bias[256 + hcol];
        const float bg = bias[512 + hcol];
        const float bo = bias[768 + hcol];
#pragma unroll
        for (int rb = 0; rb < 2; ++rb) {
#pragma unroll
            for (int r = 0; r < 8; ++r) {
                const int m = rb * 16 + rhalf + r;
                const size_t gi = (size_t)(rowbase + m) * Hsz + hcol;
                const float zi = acc[rb][0 + pp][r] + bi;
                const float zf = acc[rb][2 + pp][r] + bf;
                const float zg = acc[rb][4 + pp][r] + bg;
                const float zo = acc[rb][6 + pp][r] + bo;
                const float cn = sigmoidf_(zf) * cbuf[gi] + sigmoidf_(zi) * tanhf_(zg);
                const float hn = sigmoidf_(zo) * tanhf_(cn);
                cbuf[gi] = cn;
                const unsigned short hb = bf16bits(hn);
                hbuf_g[gi] = hb;
                if (hlds) hlds[m * Hsz + hcol] = hb;
            }
        }
    }
}

template <bool IS_ENC>
__global__ __launch_bounds__(256, 1)
void lstm_step_kernel(const float* __restrict__ x,         // enc: [B,S,D]
                      const unsigned short* inp,           // dec: [B,H] bf16 (may alias h2)
                      const unsigned short* __restrict__ Wih0,  // [G, K0]
                      const unsigned short* __restrict__ Whh0,  // [G, H]
                      const unsigned short* __restrict__ Wih1,  // [G, H]
                      const unsigned short* __restrict__ Whh1,  // [G, H]
                      const float* __restrict__ bias0,
                      const float* __restrict__ bias1,
                      unsigned short* __restrict__ h1, float* __restrict__ c1,
                      unsigned short* h2, float* __restrict__ c2,
                      const unsigned short* __restrict__ Wout,  // dec: [D, H]
                      const float* __restrict__ bout,
                      float* __restrict__ out,                  // dec: [B,S,D]
                      int t) {
    __shared__ unsigned short h1buf[BT * Hsz];   // 16 KB
    __shared__ unsigned short h2buf[BT * Hsz];   // 16 KB (decoder projection)

    const int tid     = threadIdx.x;
    const int lane    = tid & 31;
    const int w       = tid >> 5;                 // wave 0..7
    const int rowbase = blockIdx.x * BT;
    const int arow    = lane & 15;

    f32x8 acc[2][8] = {};

    // ---------------- layer 0: z0 = in @ Wih0^T + h1 @ Whh0^T ----------------
    if (IS_ENC) {
        const float* xr0 = x + (size_t)(rowbase + arow)      * (Ssz * Dsz) + (size_t)t * Dsz;
        const float* xr1 = x + (size_t)(rowbase + 16 + arow) * (Ssz * Dsz) + (size_t)t * Dsz;
#pragma unroll
        for (int kc = 0; kc < Dsz; kc += 32) {
            FragBF a0 = load_a_f32cvt(xr0, kc, lane);
            FragBF a1 = load_a_f32cvt(xr1, kc, lane);
            mm_step2(acc, a0, a1, Wih0, Dsz, kc, w, lane);
        }
    } else {
#pragma unroll
        for (int kc = 0; kc < Hsz; kc += 32) {
            FragBF a0 = load_a_bf16(inp, Hsz, rowbase + arow, kc, lane);
            FragBF a1 = load_a_bf16(inp, Hsz, rowbase + 16 + arow, kc, lane);
            mm_step2(acc, a0, a1, Wih0, Hsz, kc, w, lane);
        }
    }
#pragma unroll
    for (int kc = 0; kc < Hsz; kc += 32) {
        FragBF a0 = load_a_bf16(h1, Hsz, rowbase + arow, kc, lane);
        FragBF a1 = load_a_bf16(h1, Hsz, rowbase + 16 + arow, kc, lane);
        mm_step2(acc, a0, a1, Whh0, Hsz, kc, w, lane);
    }
    __syncthreads();   // all waves done reading h1
    cell_elemwise(acc, bias0, c1, h1, &h1buf[0], rowbase, w, lane);
    __syncthreads();   // h1buf visible

    // ---------------- layer 1: z1 = h1 @ Wih1^T + h2 @ Whh1^T ----------------
#pragma unroll
    for (int rb = 0; rb < 2; ++rb)
#pragma unroll
        for (int j = 0; j < 8; ++j) acc[rb][j] = f32x8{};
#pragma unroll
    for (int kc = 0; kc < Hsz; kc += 32) {
        FragBF a0 = load_a_bf16(&h1buf[0], Hsz, arow, kc, lane);
        FragBF a1 = load_a_bf16(&h1buf[0], Hsz, 16 + arow, kc, lane);
        mm_step2(acc, a0, a1, Wih1, Hsz, kc, w, lane);
    }
#pragma unroll
    for (int kc = 0; kc < Hsz; kc += 32) {
        FragBF a0 = load_a_bf16(h2, Hsz, rowbase + arow, kc, lane);
        FragBF a1 = load_a_bf16(h2, Hsz, rowbase + 16 + arow, kc, lane);
        mm_step2(acc, a0, a1, Whh1, Hsz, kc, w, lane);
    }
    __syncthreads();   // all waves done reading h2 (and inp, which may alias h2)
    cell_elemwise(acc, bias1, c2, h2,
                  IS_ENC ? (unsigned short*)nullptr : &h2buf[0],
                  rowbase, w, lane);

    // ---------------- decoder: fused output projection ----------------
    if (!IS_ENC) {
        __syncthreads();   // h2buf visible
        f32x8 po[2] = {};
#pragma unroll
        for (int kc = 0; kc < Hsz; kc += 32) {
            FragBF a0 = load_a_bf16(&h2buf[0], Hsz, arow, kc, lane);
            FragBF a1 = load_a_bf16(&h2buf[0], Hsz, 16 + arow, kc, lane);
            FragBF b;
            const unsigned short* p =
                Wout + (size_t)(w * 16 + arow) * Hsz + kc + ((lane >> 4) * 8);
            b.q[0] = *(const uint4*)(p);
            b.q[1] = *(const uint4*)(p + 16);
            po[0] = __builtin_amdgcn_wmma_f32_16x16x32_bf16(
                false, a0.v, false, b.v, (short)0, po[0], false, false);
            po[1] = __builtin_amdgcn_wmma_f32_16x16x32_bf16(
                false, a1.v, false, b.v, (short)0, po[1], false, false);
        }
        const float bb = bout[w * 16 + arow];
#pragma unroll
        for (int rb = 0; rb < 2; ++rb) {
#pragma unroll
            for (int r = 0; r < 8; ++r) {
                const int m = rb * 16 + (lane >> 4) * 8 + r;
                out[(size_t)(rowbase + m) * (Ssz * Dsz) + (size_t)t * Dsz +
                    w * 16 + arow] = po[rb][r] + bb;
            }
        }
    }
}

__global__ void cvt_f32_to_bf16_kernel(const float* __restrict__ s,
                                       unsigned short* __restrict__ d, int n) {
    const int i = blockIdx.x * 256 + threadIdx.x;
    if (i < n) d[i] = bf16bits(s[i]);
}

__global__ void add_bias_kernel(const float* __restrict__ a,
                                const float* __restrict__ b,
                                float* __restrict__ d, int n) {
    const int i = blockIdx.x * 256 + threadIdx.x;
    if (i < n) d[i] = a[i] + b[i];
}

extern "C" void kernel_launch(void* const* d_in, const int* in_sizes, int n_in,
                              void* d_out, int out_size, void* d_ws, size_t ws_size,
                              hipStream_t stream) {
    (void)in_sizes; (void)n_in; (void)out_size; (void)ws_size;

    const float* x     = (const float*)d_in[0];
    const float* eWih0 = (const float*)d_in[1];
    const float* eWhh0 = (const float*)d_in[2];
    const float* eBih0 = (const float*)d_in[3];
    const float* eBhh0 = (const float*)d_in[4];
    const float* eWih1 = (const float*)d_in[5];
    const float* eWhh1 = (const float*)d_in[6];
    const float* eBih1 = (const float*)d_in[7];
    const float* eBhh1 = (const float*)d_in[8];
    const float* dWih0 = (const float*)d_in[9];
    const float* dWhh0 = (const float*)d_in[10];
    const float* dBih0 = (const float*)d_in[11];
    const float* dBhh0 = (const float*)d_in[12];
    const float* dWih1 = (const float*)d_in[13];
    const float* dWhh1 = (const float*)d_in[14];
    const float* dBih1 = (const float*)d_in[15];
    const float* dBhh1 = (const float*)d_in[16];
    const float* Wout  = (const float*)d_in[17];
    const float* bout  = (const float*)d_in[18];
    float*       out   = (float*)d_out;

    // ---- carve workspace (~16 MB) ----
    char*  base = (char*)d_ws;
    size_t off  = 0;
    auto carve = [&](size_t bytes) -> char* {
        char* p = base + off;
        off += (bytes + 255) & ~(size_t)255;
        return p;
    };
    unsigned short* wih0e = (unsigned short*)carve((size_t)Gsz * Dsz * 2);
    unsigned short* whh0e = (unsigned short*)carve((size_t)Gsz * Hsz * 2);
    unsigned short* wih1e = (unsigned short*)carve((size_t)Gsz * Hsz * 2);
    unsigned short* whh1e = (unsigned short*)carve((size_t)Gsz * Hsz * 2);
    unsigned short* wih0d = (unsigned short*)carve((size_t)Gsz * Hsz * 2);
    unsigned short* whh0d = (unsigned short*)carve((size_t)Gsz * Hsz * 2);
    unsigned short* wih1d = (unsigned short*)carve((size_t)Gsz * Hsz * 2);
    unsigned short* whh1d = (unsigned short*)carve((size_t)Gsz * Hsz * 2);
    unsigned short* woutb = (unsigned short*)carve((size_t)Dsz * Hsz * 2);
    float* bias0e = (float*)carve(Gsz * 4);
    float* bias1e = (float*)carve(Gsz * 4);
    float* bias0d = (float*)carve(Gsz * 4);
    float* bias1d = (float*)carve(Gsz * 4);
    unsigned short* h1e = (unsigned short*)carve((size_t)Bsz * Hsz * 2);
    unsigned short* h2e = (unsigned short*)carve((size_t)Bsz * Hsz * 2);
    unsigned short* h1d = (unsigned short*)carve((size_t)Bsz * Hsz * 2);
    unsigned short* h2d = (unsigned short*)carve((size_t)Bsz * Hsz * 2);
    float* c1e = (float*)carve((size_t)Bsz * Hsz * 4);
    float* c2e = (float*)carve((size_t)Bsz * Hsz * 4);
    float* c1d = (float*)carve((size_t)Bsz * Hsz * 4);
    float* c2d = (float*)carve((size_t)Bsz * Hsz * 4);

    // ---- weight / bias preparation ----
    auto cvt = [&](const float* s, unsigned short* d, int n) {
        cvt_f32_to_bf16_kernel<<<(n + 255) / 256, 256, 0, stream>>>(s, d, n);
    };
    cvt(eWih0, wih0e, Gsz * Dsz);
    cvt(eWhh0, whh0e, Gsz * Hsz);
    cvt(eWih1, wih1e, Gsz * Hsz);
    cvt(eWhh1, whh1e, Gsz * Hsz);
    cvt(dWih0, wih0d, Gsz * Hsz);
    cvt(dWhh0, whh0d, Gsz * Hsz);
    cvt(dWih1, wih1d, Gsz * Hsz);
    cvt(dWhh1, whh1d, Gsz * Hsz);
    cvt(Wout,  woutb, Dsz * Hsz);
    add_bias_kernel<<<4, 256, 0, stream>>>(eBih0, eBhh0, bias0e, Gsz);
    add_bias_kernel<<<4, 256, 0, stream>>>(eBih1, eBhh1, bias1e, Gsz);
    add_bias_kernel<<<4, 256, 0, stream>>>(dBih0, dBhh0, bias0d, Gsz);
    add_bias_kernel<<<4, 256, 0, stream>>>(dBih1, dBhh1, bias1d, Gsz);

    // ---- encoder ----
    hipMemsetAsync(h1e, 0, (size_t)Bsz * Hsz * 2, stream);
    hipMemsetAsync(h2e, 0, (size_t)Bsz * Hsz * 2, stream);
    hipMemsetAsync(c1e, 0, (size_t)Bsz * Hsz * 4, stream);
    hipMemsetAsync(c2e, 0, (size_t)Bsz * Hsz * 4, stream);
    for (int t = 0; t < Ssz; ++t) {
        lstm_step_kernel<true><<<Bsz / BT, 256, 0, stream>>>(
            x, nullptr, wih0e, whh0e, wih1e, whh1e, bias0e, bias1e,
            h1e, c1e, h2e, c2e, nullptr, nullptr, nullptr, t);
    }

    // ---- decoder (autoregressive; step 0 input = encoder's final h2) ----
    hipMemsetAsync(h1d, 0, (size_t)Bsz * Hsz * 2, stream);
    hipMemsetAsync(h2d, 0, (size_t)Bsz * Hsz * 2, stream);
    hipMemsetAsync(c1d, 0, (size_t)Bsz * Hsz * 4, stream);
    hipMemsetAsync(c2d, 0, (size_t)Bsz * Hsz * 4, stream);
    for (int t = 0; t < Ssz; ++t) {
        lstm_step_kernel<false><<<Bsz / BT, 256, 0, stream>>>(
            nullptr, (t == 0) ? h2e : h2d, wih0d, whh0d, wih1d, whh1d,
            bias0d, bias1d, h1d, c1d, h2d, c2d, woutb, bout, out, t);
    }
}
